// ConvLSTMCell_86165633893135
// MI455X (gfx1250) — matrix operand order repaired and checked by reading, not compiled, not run on hardware
//
#include <hip/hip_runtime.h>

#define DD 256
#define CHOUT 128          // 4 gates x 32 channels
#define RED 432            // 3*48 (x taps) + 3*96 (h taps)
#define COUT_OFF 33554432  // 16*32*256*256

typedef __attribute__((ext_vector_type(16))) __bf16 v16bf;
typedef __attribute__((ext_vector_type(8)))  __bf16 v8bf;
typedef __attribute__((ext_vector_type(8)))  float  v8f;

union FragBF { v16bf v; v8bf h[2]; };

static __device__ __forceinline__ unsigned short f2bf(float f) {
    unsigned int u = __float_as_uint(f);
    u = (u + 0x7FFFu + ((u >> 16) & 1u)) >> 16;   // round-to-nearest-even
    return (unsigned short)u;
}

static __device__ __forceinline__ float sigmoidf(float x) {
    return 1.0f / (1.0f + __expf(-x));
}

// ---------------------------------------------------------------------------
// Weight packing: 8 conv kernels -> one bf16 [128 out][432 red] matrix + bias
// red index: r<144: x-part (dy*48 + dx*16 + cin); r>=144: h-part (dy*96+dx*32+cin)
// out index: o = gate*32 + k, gate order i,f,c,o
// ---------------------------------------------------------------------------
struct PackArgs {
    const float* Wx[4]; const float* bx[4];
    const float* Wh[4]; const float* bh[4];
};

__global__ __launch_bounds__(432) void pack_weights(PackArgs pa,
                                                    unsigned short* __restrict__ wPack,
                                                    float* __restrict__ biasPack) {
    const int o = blockIdx.x;          // 0..127
    const int r = threadIdx.x;         // 0..431
    const int g = o >> 5, k = o & 31;
    float val;
    if (r < 144) {
        int dy = r / 48, rem = r % 48, dx = rem >> 4, ci = rem & 15;
        val = pa.Wx[g][((k * 16 + ci) * 3 + dy) * 3 + dx];
    } else {
        int r2 = r - 144;
        int dy = r2 / 96, rem = r2 % 96, dx = rem >> 5, ci = rem & 31;
        val = pa.Wh[g][((k * 32 + ci) * 3 + dy) * 3 + dx];
    }
    wPack[o * RED + r] = f2bf(val);
    if (r == 0) biasPack[o] = pa.bx[g][k] + pa.bh[g][k];
}

// ---------------------------------------------------------------------------
// Fused ConvLSTM cell: BN -> 8 convs as one implicit GEMM (WMMA bf16) ->
// gate math + peephole -> h_new, c_new.  One WG = 128-pixel row segment.
// ---------------------------------------------------------------------------
__global__ __launch_bounds__(256) void convlstm_main(
    const float* __restrict__ x, const float* __restrict__ hin,
    const float* __restrict__ cin_g,
    const float* __restrict__ bn_g, const float* __restrict__ bn_b,
    const float* __restrict__ bn_m, const float* __restrict__ bn_v,
    const unsigned short* __restrict__ wPack, const float* __restrict__ biasPack,
    const float* __restrict__ Wci, const float* __restrict__ Wcf,
    const float* __restrict__ Wco, float* __restrict__ out)
{
    __shared__ __align__(16) unsigned short wLDS[CHOUT * RED];  // 110592 B
    __shared__ __align__(16) unsigned short actLDS[18720];      //  37440 B

    const int xseg = blockIdx.x;          // 0..1
    const int y    = blockIdx.y;          // 0..255
    const int b    = blockIdx.z;          // 0..15
    const int x0   = xseg * 128;
    const int tid  = threadIdx.x;

    // ---- prefetch elementwise-phase operands (c + peepholes) while we GEMM ----
    {
        const int k     = (tid >> 3) & 31;   // 32 channels x 8 x 16-px chunks = 256
        const int seg16 = tid & 7;
        const int off   = (k * DD + y) * DD + x0 + seg16 * 16;
        __builtin_prefetch(cin_g + (b * 32 * DD * DD) + off, 0, 1);
        __builtin_prefetch(Wci + off, 0, 1);
        __builtin_prefetch(Wcf + off, 0, 1);
        __builtin_prefetch(Wco + off, 0, 1);
    }

    // ---- stage packed weights (contiguous b128 copies) ----
    {
        const uint4* src = (const uint4*)wPack;
        uint4* dst = (uint4*)wLDS;
        #pragma unroll 4
        for (int i = tid; i < (CHOUT * RED) / 8; i += 256) dst[i] = src[i];
    }

    // ---- stage BN(x) halo tile: [dy 0..2][col 0..129][cin 0..15] ----
    for (int i = tid; i < 3 * 16 * 130; i += 256) {
        int col = i % 130, t = i / 130, ci = t & 15, dy = t >> 4;
        int yy = y - 1 + dy, px = x0 - 1 + col;
        unsigned short bfv = 0;  // SAME padding pads the *normalized* input with 0
        if (yy >= 0 && yy < DD && px >= 0 && px < DD) {
            float v = x[((b * 16 + ci) * DD + yy) * DD + px];
            float inv = rsqrtf(bn_v[ci] + 1e-5f);
            v = (v - bn_m[ci]) * (bn_g[ci] * inv) + bn_b[ci];
            bfv = f2bf(v);
        }
        actLDS[(dy * 130 + col) * 16 + ci] = bfv;
    }

    // ---- stage h halo tile: [dy][col][cin 0..31] at half-offset 6240 ----
    for (int i = tid; i < 3 * 32 * 130; i += 256) {
        int col = i % 130, t = i / 130, ci = t & 31, dy = t >> 5;
        int yy = y - 1 + dy, px = x0 - 1 + col;
        unsigned short bfv = 0;
        if (yy >= 0 && yy < DD && px >= 0 && px < DD)
            bfv = f2bf(hin[((b * 32 + ci) * DD + yy) * DD + px]);
        actLDS[6240 + (dy * 130 + col) * 32 + ci] = bfv;
    }
    __syncthreads();

    // ---- implicit GEMM: wave w owns pixels [16w,16w+16), all 128 channels ----
    const int lane = tid & 31;
    const int wave = tid >> 5;
    const int ln   = lane & 15;
    const int mloc = wave * 16 + ln;               // A-row (pixel) for this lane
    const int g0   = (lane < 16) ? 0 : 8;          // A fragment K-group base
    const int kbB  = (lane < 16) ? 0 : 16;         // unused; B fragment loads also use g0

    v8bf z8;
    #pragma unroll
    for (int i = 0; i < 8; ++i) z8[i] = (__bf16)0.0f;

    v8f acc[8];
    #pragma unroll
    for (int t = 0; t < 8; ++t)
        #pragma unroll
        for (int e = 0; e < 8; ++e) acc[t][e] = 0.0f;

    const __bf16* actB     = (const __bf16*)actLDS;
    const __bf16* wRowBase = (const __bf16*)wLDS + ln * RED;

    // Straight-line GEMM chunk:
    //  - full chunk (kval=32): load both A groups; all K slots real.
    //  - partial chunk (kval=16): A group1 (covers all K>=16 slots) is forced to
    //    zero at COMPILE TIME, so B can be loaded unconditionally — garbage K>=16
    //    columns of B multiply by zero. No EXEC masking, no selects.
    #define GEMM_CHUNK(aPtr, wOff, PARTIAL)                                       \
    {                                                                             \
        FragBF af;                                                                \
        af.h[0] = *(const v8bf*)((aPtr) + g0);                                    \
        af.h[1] = (PARTIAL) ? z8 : *(const v8bf*)((aPtr) + g0 + 16);              \
        _Pragma("unroll")                                                         \
        for (int t = 0; t < 8; ++t) {                                             \
            const __bf16* wr = wRowBase + t * (16 * RED) + (wOff);                \
            FragBF bfr;                                                           \
            bfr.h[0] = *(const v8bf*)(wr + g0);                                   \
            bfr.h[1] = *(const v8bf*)(wr + g0 + 16);                              \
            acc[t] = __builtin_amdgcn_wmma_f32_16x16x32_bf16(                     \
                false, af.v, false, bfr.v, (short)0, acc[t], false, false);       \
        }                                                                         \
    }

    // reduction segments: 3 x-rows (len 48, stride 16), 3 h-rows (len 96, stride 32)
    {
        constexpr int segAct[6]    = {0, 2080, 4160, 6240, 10400, 14560};
        constexpr int segStride[6] = {16, 16, 16, 32, 32, 32};
        constexpr int segLen[6]    = {48, 48, 48, 96, 96, 96};
        constexpr int segW[6]      = {0, 48, 96, 144, 240, 336};
        #pragma unroll
        for (int s = 0; s < 6; ++s) {
            const __bf16* aBase = actB + segAct[s] + mloc * segStride[s];
            #pragma unroll
            for (int k0 = 0; k0 < segLen[s]; k0 += 32) {
                if (segLen[s] - k0 >= 32) {
                    GEMM_CHUNK(aBase + k0, segW[s] + k0, false)
                } else {
                    GEMM_CHUNK(aBase + k0, segW[s] + k0, true)
                }
            }
        }
    }
    #undef GEMM_CHUNK

    // ---- gate math in registers: channel k lives in tiles {t0, t0+2, t0+4, t0+6} ----
    #pragma unroll
    for (int t0 = 0; t0 < 2; ++t0) {
        const int k = ln + 16 * t0;
        const float bi = biasPack[k];
        const float bf = biasPack[32 + k];
        const float bc = biasPack[64 + k];
        const float bo = biasPack[96 + k];
        #pragma unroll
        for (int v = 0; v < 8; ++v) {
            const int m  = wave * 16 + (lane >> 4) * 8 + v;   // D layout: M = v + 8*(lane/16)
            const int px = x0 + m;
            const int idx  = ((b * 32 + k) * DD + y) * DD + px;
            const int pidx = (k * DD + y) * DD + px;
            const float cprev = cin_g[idx];
            const float a_i = acc[t0    ][v] + bi + cprev * Wci[pidx];
            const float a_f = acc[t0 + 2][v] + bf + cprev * Wcf[pidx];
            const float a_c = acc[t0 + 4][v] + bc;
            const float a_o = acc[t0 + 6][v] + bo;
            const float ig = sigmoidf(a_i);
            const float fg = sigmoidf(a_f);
            const float cn = fg * cprev + ig * tanhf(a_c);
            const float og = sigmoidf(a_o + cn * Wco[pidx]);
            out[idx]            = og * tanhf(cn);   // h_new
            out[COUT_OFF + idx] = cn;               // c_new
        }
    }
}

extern "C" void kernel_launch(void* const* d_in, const int* in_sizes, int n_in,
                              void* d_out, int out_size, void* d_ws, size_t ws_size,
                              hipStream_t stream) {
    (void)in_sizes; (void)n_in; (void)out_size; (void)ws_size;
    const float* x    = (const float*)d_in[0];
    const float* h    = (const float*)d_in[1];
    const float* c    = (const float*)d_in[2];
    const float* bn_g = (const float*)d_in[3];
    const float* bn_b = (const float*)d_in[4];
    const float* bn_m = (const float*)d_in[5];
    const float* bn_v = (const float*)d_in[6];

    PackArgs pa;
    for (int g = 0; g < 4; ++g) {
        pa.Wx[g] = (const float*)d_in[7 + 2 * g];
        pa.bx[g] = (const float*)d_in[8 + 2 * g];
        pa.Wh[g] = (const float*)d_in[15 + 2 * g];
        pa.bh[g] = (const float*)d_in[16 + 2 * g];
    }
    const float* Wci = (const float*)d_in[23];
    const float* Wcf = (const float*)d_in[24];
    const float* Wco = (const float*)d_in[25];

    unsigned short* wPack = (unsigned short*)d_ws;
    float* biasPack = (float*)((char*)d_ws + (size_t)CHOUT * RED * 2);

    pack_weights<<<CHOUT, RED, 0, stream>>>(pa, wPack, biasPack);

    dim3 grid(2, DD, 16);   // (x-segment, y, batch)
    convlstm_main<<<grid, 256, 0, stream>>>(x, h, c, bn_g, bn_b, bn_m, bn_v,
                                            wPack, biasPack, Wci, Wcf, Wco,
                                            (float*)d_out);
}
